// MultiHeadAttention_67628555043327
// MI455X (gfx1250) — matrix operand order repaired and checked by reading, not compiled, not run on hardware
//
#include <hip/hip_runtime.h>
#include <hip/hip_bf16.h>

// ---------------------------------------------------------------------------
// Problem constants (from reference): D_MODEL=1024, H=16, Dk=64, B=2, S=2048
// ---------------------------------------------------------------------------
#define DMODEL 1024
#define NHEADS 16
#define DK     64
#define BATCH  2
#define SEQ    2048
#define MTOT   (BATCH * SEQ)   // 4096 rows for the projections

typedef __attribute__((ext_vector_type(16))) __bf16 v16bf;
typedef __attribute__((ext_vector_type(8)))  __bf16 v8bf;
typedef __attribute__((ext_vector_type(4)))  __bf16 v4bf;
typedef __attribute__((ext_vector_type(8)))  float  v8f;

union ABfrag { v16bf v; v8bf h[2]; };

__device__ inline __bf16 f2bf(float x) {
    unsigned u = __builtin_bit_cast(unsigned, x);
    u += 0x7FFFu + ((u >> 16) & 1u);          // round-to-nearest-even
    unsigned short hs = (unsigned short)(u >> 16);
    return __builtin_bit_cast(__bf16, hs);
}

__device__ inline v8f vzero8() {
    v8f z;
#pragma unroll
    for (int i = 0; i < 8; ++i) z[i] = 0.0f;
    return z;
}

__device__ inline v8f wmma_bf16(const v16bf& a, const v16bf& b, const v8f& c) {
    return __builtin_amdgcn_wmma_f32_16x16x32_bf16(false, a, false, b,
                                                   (short)0, c, false, false);
}

// ---- 16-lane xor reductions in VALU via DPP16 ROW_XMASK (no LDS traffic) ----
template <int CTRL>
__device__ inline float dpp_xor(float x) {
    int r = __builtin_amdgcn_update_dpp(0, __builtin_bit_cast(int, x),
                                        CTRL, 0xf, 0xf, true);
    return __builtin_bit_cast(float, r);
}
__device__ inline float rowmax16(float x) {
    x = fmaxf(x, dpp_xor<0x161>(x));   // ROW_XMASK ^1
    x = fmaxf(x, dpp_xor<0x162>(x));   // ROW_XMASK ^2
    x = fmaxf(x, dpp_xor<0x164>(x));   // ROW_XMASK ^4
    x = fmaxf(x, dpp_xor<0x168>(x));   // ROW_XMASK ^8
    return x;
}
__device__ inline float rowsum16(float x) {
    x += dpp_xor<0x161>(x);
    x += dpp_xor<0x162>(x);
    x += dpp_xor<0x164>(x);
    x += dpp_xor<0x168>(x);
    return x;
}

// ---------------------------------------------------------------------------
// GEMM: C[M,N] = A[M,K] @ W[N,K]^T + bias  (torch Linear)
//   MODE 0: A is f32 (activation), output bf16 scattered to [B,H,S,Dk]
//   MODE 1: A is bf16 (ctx),       output f32 row-major [M,N]
// Block tile 128x128, K-step 32, 8 waves in 4x2 (each wave: 32x64 = 2x4 WMMA)
// ---------------------------------------------------------------------------
#define BM 128
#define BN 128
#define BK 32
#define LDA 40   // halves per Atile row (80B, 16B aligned)
#define LDB 40   // halves per Btile row

template <int MODE>
__global__ __launch_bounds__(256) void gemm_wmma(const void* __restrict__ Aptr,
                                                 const float* __restrict__ W,
                                                 const float* __restrict__ bias,
                                                 void* __restrict__ outp) {
    __shared__ __bf16 Atile[BM * LDA];
    __shared__ __bf16 Btile[BN * LDB];

    const int K = DMODEL, N = DMODEL;
    const int tid  = threadIdx.x;
    const int lane = tid & 31;
    const int wave = tid >> 5;
    const int wm   = wave & 3;   // 4 waves along M (32 rows each)
    const int wn   = wave >> 2;  // 2 waves along N (64 cols each)
    const int hi   = lane >> 4;
    const int lo   = lane & 15;
    const int m0 = blockIdx.x * BM;
    const int n0 = blockIdx.y * BN;

    const float*  Af32 = (const float*)Aptr;
    const __bf16* Abf  = (const __bf16*)Aptr;

    v8f c[2][4];
#pragma unroll
    for (int i = 0; i < 2; ++i)
#pragma unroll
        for (int j = 0; j < 4; ++j) c[i][j] = vzero8();

    for (int k0 = 0; k0 < K; k0 += BK) {
        // ---- stage A tile (128x32) as bf16, vectorized ----
        if (MODE == 0) {
#pragma unroll
            for (int i = 0; i < 4; ++i) {           // 1024 float4 groups
                int idx = tid + i * 256;
                int r = idx >> 3, q = idx & 7;
                const float4 f = *(const float4*)&Af32[(size_t)(m0 + r) * K + k0 + 4 * q];
                v4bf p; p[0] = f2bf(f.x); p[1] = f2bf(f.y); p[2] = f2bf(f.z); p[3] = f2bf(f.w);
                *(v4bf*)&Atile[r * LDA + 4 * q] = p;
            }
        } else {
#pragma unroll
            for (int i = 0; i < 2; ++i) {           // 512 v8bf groups
                int idx = tid + i * 256;
                int r = idx >> 2, q = idx & 3;
                *(v8bf*)&Atile[r * LDA + 8 * q] =
                    *(const v8bf*)&Abf[(size_t)(m0 + r) * K + k0 + 8 * q];
            }
        }
        // ---- stage B tile: Bt[n][k] = bf16(W[n0+n][k0+k]), vectorized ----
#pragma unroll
        for (int i = 0; i < 4; ++i) {               // 1024 float4 groups
            int idx = tid + i * 256;
            int nn = idx >> 3, q = idx & 7;
            const float4 f = *(const float4*)&W[(size_t)(n0 + nn) * K + k0 + 4 * q];
            v4bf p; p[0] = f2bf(f.x); p[1] = f2bf(f.y); p[2] = f2bf(f.z); p[3] = f2bf(f.w);
            *(v4bf*)&Btile[nn * LDB + 4 * q] = p;
        }
        __syncthreads();

        ABfrag a[2], b[4];
#pragma unroll
        for (int i = 0; i < 2; ++i) {
            int row = wm * 32 + i * 16 + lo;
            a[i].h[0] = *(const v8bf*)&Atile[row * LDA + hi * 8];
            a[i].h[1] = *(const v8bf*)&Atile[row * LDA + 16 + hi * 8];
        }
#pragma unroll
        for (int j = 0; j < 4; ++j) {
            int col = wn * 64 + j * 16 + lo;
            b[j].h[0] = *(const v8bf*)&Btile[col * LDB + hi * 8];
            b[j].h[1] = *(const v8bf*)&Btile[col * LDB + 16 + hi * 8];
        }
#pragma unroll
        for (int i = 0; i < 2; ++i)
#pragma unroll
            for (int j = 0; j < 4; ++j)
                c[i][j] = wmma_bf16(a[i].v, b[j].v, c[i][j]);
        __syncthreads();
    }

    // ---- epilogue ----
#pragma unroll
    for (int i = 0; i < 2; ++i) {
#pragma unroll
        for (int j = 0; j < 4; ++j) {
            int n  = n0 + wn * 64 + j * 16 + lo;
            int mb = m0 + wm * 32 + i * 16 + hi * 8;
            float bv = bias[n];
#pragma unroll
            for (int v = 0; v < 8; ++v) {
                int m = mb + v;
                float val = c[i][j][v] + bv;
                if (MODE == 0) {
                    int bb = m >> 11, ss = m & (SEQ - 1);
                    int hh = n >> 6,  dk = n & (DK - 1);
                    size_t off = (((size_t)(bb * NHEADS + hh) * SEQ) + ss) * DK + dk;
                    ((__bf16*)outp)[off] = f2bf(val);
                } else {
                    ((float*)outp)[(size_t)m * DMODEL + n] = val;
                }
            }
        }
    }
}

// ---------------------------------------------------------------------------
// Flash attention: grid (S/128, B*H); 8 waves x 16 queries; keys in 32-chunks.
// Qh/Kh/Vh layout [B,H,S,Dk] bf16.  ctx out [B,S,H*Dk] bf16.
// Softmax tracked in exp2 domain: s' = s * (1/8 * log2 e).
// ---------------------------------------------------------------------------
#define LDK 72   // Ktile[key][d]  (144B rows, 16B aligned)
#define LDV 40   // Vt[d][key]     (80B rows)
#define LDP 40   // per-wave P scratch rows
#define SM_SCALE 0.18033688011112042f   // 0.125 * log2(e)

__global__ __launch_bounds__(256) void attn_flash(const __bf16* __restrict__ Qh,
                                                  const __bf16* __restrict__ Kh,
                                                  const __bf16* __restrict__ Vh,
                                                  __bf16* __restrict__ ctx) {
    __shared__ __bf16 Kt[32 * LDK];
    __shared__ __bf16 Vt[DK * LDV];
    __shared__ __bf16 Pl[8 * 16 * LDP];

    const int tid  = threadIdx.x;
    const int lane = tid & 31;
    const int wave = tid >> 5;
    const int hi   = lane >> 4;
    const int lo   = lane & 15;
    const int bh   = blockIdx.y;            // b*H + h
    const int q0   = blockIdx.x * 128 + wave * 16;

    const __bf16* Qb = Qh + (size_t)bh * SEQ * DK;
    const __bf16* Kb = Kh + (size_t)bh * SEQ * DK;
    const __bf16* Vb = Vh + (size_t)bh * SEQ * DK;

    // load Q fragments (16 queries x 64 d = two K=32 A-frags), kept in regs
    ABfrag qa[2];
    {
        int qrow = q0 + lo;
#pragma unroll
        for (int d0 = 0; d0 < 2; ++d0) {
            qa[d0].h[0] = *(const v8bf*)&Qb[(size_t)qrow * DK + d0 * 32 + hi * 8];
            qa[d0].h[1] = *(const v8bf*)&Qb[(size_t)qrow * DK + d0 * 32 + 16 + hi * 8];
        }
    }

    float mrow[8], lrow[8];
    v8f acc[4];
#pragma unroll
    for (int v = 0; v < 8; ++v) { mrow[v] = -3.0e38f; lrow[v] = 0.0f; }
#pragma unroll
    for (int t = 0; t < 4; ++t) acc[t] = vzero8();

    __bf16* Pw = Pl + wave * 16 * LDP;

    // staging coordinates: 256 threads == 32 keys x 8 v8bf groups
    const int skey = tid >> 3;
    const int sg   = tid & 7;

    for (int kb = 0; kb < SEQ / 32; ++kb) {
        const int kk0 = kb * 32;
        __syncthreads();
        // K tile [32 keys][64 d]: b128 load -> b128 store
        *(v8bf*)&Kt[skey * LDK + sg * 8] =
            *(const v8bf*)&Kb[(size_t)(kk0 + skey) * DK + sg * 8];
        // V transposed [64 d][32 keys]: b128 load -> 8 b16 scatter stores
        {
            v8bf vv = *(const v8bf*)&Vb[(size_t)(kk0 + skey) * DK + sg * 8];
#pragma unroll
            for (int e = 0; e < 8; ++e) Vt[(sg * 8 + e) * LDV + skey] = vv[e];
        }
        __syncthreads();

        // scores: 16 queries x 32 keys (two 16x16 C tiles, d reduced over 64)
        v8f sc[2];
#pragma unroll
        for (int tj = 0; tj < 2; ++tj) {
            ABfrag kf[2];
            int keyl = tj * 16 + lo;
#pragma unroll
            for (int d0 = 0; d0 < 2; ++d0) {
                kf[d0].h[0] = *(const v8bf*)&Kt[keyl * LDK + d0 * 32 + hi * 8];
                kf[d0].h[1] = *(const v8bf*)&Kt[keyl * LDK + d0 * 32 + 16 + hi * 8];
            }
            v8f s = vzero8();
            s = wmma_bf16(qa[0].v, kf[0].v, s);
            s = wmma_bf16(qa[1].v, kf[1].v, s);
            sc[tj] = s;
        }

        // online softmax in exp2 domain; row reductions via DPP (VALU only)
#pragma unroll
        for (int v = 0; v < 8; ++v) {
            float s0 = sc[0][v] * SM_SCALE;
            float s1 = sc[1][v] * SM_SCALE;
            float mx = rowmax16(fmaxf(s0, s1));
            float mn = fmaxf(mrow[v], mx);
            float scale = __builtin_amdgcn_exp2f(mrow[v] - mn);
            float p0 = __builtin_amdgcn_exp2f(s0 - mn);
            float p1 = __builtin_amdgcn_exp2f(s1 - mn);
            sc[0][v] = p0; sc[1][v] = p1;
            float sum = rowsum16(p0 + p1);
            lrow[v] = lrow[v] * scale + sum;
            mrow[v] = mn;
#pragma unroll
            for (int t = 0; t < 4; ++t) acc[t][v] *= scale;
        }

        // transpose P (C layout -> A layout) through per-wave LDS scratch
#pragma unroll
        for (int tj = 0; tj < 2; ++tj)
#pragma unroll
            for (int v = 0; v < 8; ++v)
                Pw[(hi * 8 + v) * LDP + tj * 16 + lo] = f2bf(sc[tj][v]);

        ABfrag pa;
        pa.h[0] = *(const v8bf*)&Pw[lo * LDP + hi * 8];
        pa.h[1] = *(const v8bf*)&Pw[lo * LDP + 16 + hi * 8];

        // O += P (16x32) @ V (32x64), four N=16 tiles
#pragma unroll
        for (int t = 0; t < 4; ++t) {
            ABfrag vf;
            int drow = t * 16 + lo;
            vf.h[0] = *(const v8bf*)&Vt[drow * LDV + hi * 8];
            vf.h[1] = *(const v8bf*)&Vt[drow * LDV + 16 + hi * 8];
            acc[t] = wmma_bf16(pa.v, vf.v, acc[t]);
        }
    }

    // finalize: divide by l, scatter to ctx [B,S,H*Dk]
    const int b = bh >> 4, h = bh & (NHEADS - 1);
#pragma unroll
    for (int t = 0; t < 4; ++t) {
#pragma unroll
        for (int v = 0; v < 8; ++v) {
            float o = acc[t][v] / lrow[v];
            int q = q0 + hi * 8 + v;
            int d = t * 16 + lo;
            ctx[((size_t)(b * SEQ + q)) * DMODEL + h * DK + d] = f2bf(o);
        }
    }
}

// ---------------------------------------------------------------------------
// Residual add + LayerNorm: one block per row (1024 cols, 256 threads x 4)
// ---------------------------------------------------------------------------
__global__ __launch_bounds__(256) void ln_res(const float* __restrict__ y,
                                              const float* __restrict__ resid,
                                              const float* __restrict__ gamma,
                                              const float* __restrict__ beta,
                                              float* __restrict__ out) {
    __shared__ float red[2][8];
    const int row = blockIdx.x;
    const int tid = threadIdx.x;
    const int lane = tid & 31, wave = tid >> 5;
    const size_t base = (size_t)row * DMODEL;

    float x[4], s = 0.0f, s2 = 0.0f;
#pragma unroll
    for (int i = 0; i < 4; ++i) {
        int c0 = tid + i * 256;
        float xv = y[base + c0] + resid[base + c0];
        x[i] = xv;
        s += xv; s2 += xv * xv;
    }
#pragma unroll
    for (int msk = 1; msk <= 16; msk <<= 1) {
        s  += __shfl_xor(s,  msk, 32);
        s2 += __shfl_xor(s2, msk, 32);
    }
    if (lane == 0) { red[0][wave] = s; red[1][wave] = s2; }
    __syncthreads();
    float ts = 0.0f, ts2 = 0.0f;
#pragma unroll
    for (int w = 0; w < 8; ++w) { ts += red[0][w]; ts2 += red[1][w]; }
    float mean = ts * (1.0f / DMODEL);
    float var  = ts2 * (1.0f / DMODEL) - mean * mean;
    float rstd = rsqrtf(var + 1e-5f);
#pragma unroll
    for (int i = 0; i < 4; ++i) {
        int c0 = tid + i * 256;
        out[base + c0] = (x[i] - mean) * rstd * gamma[c0] + beta[c0];
    }
}

// ---------------------------------------------------------------------------
// Launch
// ---------------------------------------------------------------------------
extern "C" void kernel_launch(void* const* d_in, const int* in_sizes, int n_in,
                              void* d_out, int out_size, void* d_ws, size_t ws_size,
                              hipStream_t stream) {
    const float* query   = (const float*)d_in[0];
    const float* key_in  = (const float*)d_in[1];
    const float* value   = (const float*)d_in[2];
    const float* w_q     = (const float*)d_in[3];
    const float* b_q     = (const float*)d_in[4];
    const float* w_k     = (const float*)d_in[5];
    const float* b_k     = (const float*)d_in[6];
    const float* w_v     = (const float*)d_in[7];
    const float* b_v     = (const float*)d_in[8];
    const float* w_o     = (const float*)d_in[9];
    const float* b_o     = (const float*)d_in[10];
    const float* ln_g    = (const float*)d_in[11];
    const float* ln_b    = (const float*)d_in[12];

    char* ws = (char*)d_ws;
    __bf16* Qh  = (__bf16*)(ws);                       // 8 MB
    __bf16* Kh  = (__bf16*)(ws + (size_t)(8u << 20));  // 8 MB
    __bf16* Vh  = (__bf16*)(ws + (size_t)(16u << 20)); // 8 MB
    __bf16* ctx = (__bf16*)(ws + (size_t)(24u << 20)); // 8 MB
    float*  yb  = (float*)(ws);                        // 16 MB, reuses Qh/Kh (dead by then)

    dim3 gg(MTOT / BM, DMODEL / BN);   // 32 x 8
    gemm_wmma<0><<<gg, 256, 0, stream>>>(query,  w_q, b_q, Qh);
    gemm_wmma<0><<<gg, 256, 0, stream>>>(key_in, w_k, b_k, Kh);
    gemm_wmma<0><<<gg, 256, 0, stream>>>(value,  w_v, b_v, Vh);

    attn_flash<<<dim3(SEQ / 128, BATCH * NHEADS), 256, 0, stream>>>(Qh, Kh, Vh, ctx);

    gemm_wmma<1><<<gg, 256, 0, stream>>>(ctx, w_o, b_o, yb);

    ln_res<<<MTOT, 256, 0, stream>>>(yb, query, ln_g, ln_b, (float*)d_out);
}
